// AttnDecoderGRU_15539191677208
// MI455X (gfx1250) — compile-verified, matrix operand order checked by reading, not run on hardware
//
#include <hip/hip_runtime.h>
#include <hip/hip_bf16.h>
#include <math.h>

#define B_ 64
#define S_ 64
#define T_ 32
#define H_ 1024
#define V_ 32000

typedef __bf16 bf16_t;
typedef __attribute__((ext_vector_type(16))) __bf16 v16bf;
typedef __attribute__((ext_vector_type(8)))  __bf16 v8bf;
typedef __attribute__((ext_vector_type(8)))  float  v8f;

// ---------------------------------------------------------------------------
// f32 -> bf16 conversion (grid-stride)
// ---------------------------------------------------------------------------
__global__ __launch_bounds__(256) void k_f32_to_bf16(const float* __restrict__ src,
                                                     bf16_t* __restrict__ dst, long n) {
  long i = (long)blockIdx.x * blockDim.x + threadIdx.x;
  long stride = (long)gridDim.x * blockDim.x;
  for (; i < n; i += stride) dst[i] = (bf16_t)src[i];
}

// h = encoder_hidden[0]; h_bf16 = bf16(h)
__global__ __launch_bounds__(256) void k_init_h(const float* __restrict__ h0,
                                                float* __restrict__ h,
                                                bf16_t* __restrict__ hb, int n) {
  int i = blockIdx.x * blockDim.x + threadIdx.x;
  if (i < n) { float v = h0[i]; h[i] = v; hb[i] = (bf16_t)v; }
}

// ---------------------------------------------------------------------------
// Register-blocked, double-buffered WMMA GEMM:
//   C[m, n] = sum_k A[m,k] * W[n,k] + bias[n]
//   A : bf16 [M x K] row-major,  W : bf16 [N x K] row-major (B = W^T)
//   C : f32, element (m,n) at C[m*ldc + n]
// One wave owns a 64(M) x 16(N) strip: 4 accumulators share each B tile
// (1x weight traffic).  K-steps are double-buffered: the WMMAs of step n
// consume registers loaded during step n-1 while step n's loads (for n+1)
// are in flight, so the wait before the WMMA group is partial, not a full
// drain.  M must be a multiple of 64 (here M = 64 or 4096); K >= 64.
// Per-lane packing follows ISA 7.12.2 (16-bit A 16x32): lanes 0-15 hold
// K {0..7, 16..23}, lanes 16-31 hold K {8..15, 24..31}; the B operand uses
// the identical packing applied to rows of W (columns of W^T).
// ---------------------------------------------------------------------------
__global__ __launch_bounds__(256) void k_wmma_gemm(const bf16_t* __restrict__ A,
                                                   const bf16_t* __restrict__ W,
                                                   const float* __restrict__ bias,
                                                   float* __restrict__ C,
                                                   int M, int N, int K, long ldc) {
  const int lane   = threadIdx.x & 31;
  const int wave   = threadIdx.x >> 5;
  const int tileId = blockIdx.x * 8 + wave;
  const int tilesN = N >> 4;
  const int tMg = tileId / tilesN;          // group of 4 M-tiles (64 rows)
  const int tN  = tileId % tilesN;
  if (tMg * 64 >= M) return;

  const int  r16   = lane & 15;
  const long kHalf = (lane >> 4) << 3;      // 0 or 8
  const bf16_t* wRow = W + (long)(tN * 16 + r16) * K + kHalf;
  const bf16_t* aRow = A + (long)(tMg * 64 + r16) * K + kHalf;
  const long aStep = (long)16 * K;          // one M-tile down

  v8f acc[4] = {{}, {}, {}, {}};

  // ---- prologue: stage K-step 0 -------------------------------------------
  v8bf wlo = *(const v8bf*)(wRow);
  v8bf whi = *(const v8bf*)(wRow + 16);
  v8bf alo[4], ahi[4];
#pragma unroll
  for (int i = 0; i < 4; ++i) {
    alo[i] = *(const v8bf*)(aRow + i * aStep);
    ahi[i] = *(const v8bf*)(aRow + i * aStep + 16);
  }

  // ---- steady state: load step k0+32 while computing step k0 --------------
  for (int k0 = 0; k0 < K - 32; k0 += 32) {
    const int kn = k0 + 32;
    v8bf wloN = *(const v8bf*)(wRow + kn);
    v8bf whiN = *(const v8bf*)(wRow + kn + 16);
    v8bf aloN[4], ahiN[4];
#pragma unroll
    for (int i = 0; i < 4; ++i) {
      aloN[i] = *(const v8bf*)(aRow + i * aStep + kn);
      ahiN[i] = *(const v8bf*)(aRow + i * aStep + kn + 16);
    }

    v16bf b;
#pragma unroll
    for (int e = 0; e < 8; ++e) { b[e] = wlo[e]; b[e + 8] = whi[e]; }
#pragma unroll
    for (int i = 0; i < 4; ++i) {
      v16bf a;
#pragma unroll
      for (int e = 0; e < 8; ++e) { a[e] = alo[i][e]; a[e + 8] = ahi[i][e]; }
      acc[i] = __builtin_amdgcn_wmma_f32_16x16x32_bf16(false, a, false, b,
                                                       (short)0, acc[i],
                                                       false, false);
    }

    wlo = wloN; whi = whiN;
#pragma unroll
    for (int i = 0; i < 4; ++i) { alo[i] = aloN[i]; ahi[i] = ahiN[i]; }
  }

  // ---- epilogue: compute last staged K-step -------------------------------
  {
    v16bf b;
#pragma unroll
    for (int e = 0; e < 8; ++e) { b[e] = wlo[e]; b[e + 8] = whi[e]; }
#pragma unroll
    for (int i = 0; i < 4; ++i) {
      v16bf a;
#pragma unroll
      for (int e = 0; e < 8; ++e) { a[e] = alo[i][e]; a[e + 8] = ahi[i][e]; }
      acc[i] = __builtin_amdgcn_wmma_f32_16x16x32_bf16(false, a, false, b,
                                                       (short)0, acc[i],
                                                       false, false);
    }
  }

  // C/D layout: VGPR r, lane l -> M = r + (l>=16 ? 8 : 0), N = l&15
  const int nOut = tN * 16 + r16;
  const float bv = bias ? bias[nOut] : 0.0f;
#pragma unroll
  for (int i = 0; i < 4; ++i) {
    const int mBase = tMg * 64 + i * 16 + ((lane >> 4) << 3);
#pragma unroll
    for (int r = 0; r < 8; ++r)
      C[(long)(mBase + r) * ldc + nOut] = acc[i][r] + bv;
  }
}

// ---------------------------------------------------------------------------
// Attention step: one block per batch element b.
//   scores[s] = Va . tanh(q[b] + keys_proj[b,s]) + b_va ; softmax over S
//   ctx = sum_s w[s] * enc[b,s,:] ; x_bf16 = [emb[tok] , ctx] ; attn out
// ---------------------------------------------------------------------------
__global__ __launch_bounds__(256) void k_attn(const float* __restrict__ q,
                                              const float* __restrict__ keys,
                                              const float* __restrict__ enc,
                                              const float* __restrict__ emb,
                                              const int*   __restrict__ tgt,
                                              const float* __restrict__ Va,
                                              const float* __restrict__ b_va,
                                              bf16_t* __restrict__ x_b,
                                              float* __restrict__ attn_out,
                                              int t) {
  const int b   = blockIdx.x;
  const int tid = threadIdx.x;
  __shared__ float qs[H_];
  __shared__ float sc[S_];
  __shared__ float w[S_];

  for (int j = tid; j < H_; j += 256) qs[j] = q[(long)b * H_ + j];
  __syncthreads();

  if (tid < S_) {
    const float* kp = keys + ((long)b * S_ + tid) * H_;
    float acc = 0.0f;
    for (int j = 0; j < H_; ++j) acc += Va[j] * tanhf(qs[j] + kp[j]);
    sc[tid] = acc + b_va[0];
  }
  __syncthreads();

  if (tid == 0) {
    float m = sc[0];
    for (int s = 1; s < S_; ++s) m = fmaxf(m, sc[s]);
    float sum = 0.0f;
    for (int s = 0; s < S_; ++s) { float e = expf(sc[s] - m); w[s] = e; sum += e; }
    float inv = 1.0f / sum;
    for (int s = 0; s < S_; ++s) w[s] *= inv;
  }
  __syncthreads();

  // context -> x[:, H:2H]
  for (int j = tid; j < H_; j += 256) {
    float c = 0.0f;
    const float* e = enc + (long)b * S_ * H_ + j;
    for (int s = 0; s < S_; ++s) c += w[s] * e[(long)s * H_];
    x_b[(long)b * 2 * H_ + H_ + j] = (bf16_t)c;
  }
  // embedding of teacher-forced token -> x[:, 0:H]   (SOS = 0 at t==0)
  const int tok = (t == 0) ? 0 : tgt[b * T_ + (t - 1)];
  for (int j = tid; j < H_; j += 256)
    x_b[(long)b * 2 * H_ + j] = (bf16_t)emb[(long)tok * H_ + j];
  // attention weights output [B,T,S]
  if (tid < S_) attn_out[((long)b * T_ + t) * S_ + tid] = w[tid];
}

// ---------------------------------------------------------------------------
// GRU elementwise update (PyTorch convention)
// ---------------------------------------------------------------------------
__device__ __forceinline__ float sigmoidf_(float x) { return 1.0f / (1.0f + expf(-x)); }

__global__ __launch_bounds__(256) void k_gru(const float* __restrict__ gi,
                                             const float* __restrict__ gh,
                                             float* __restrict__ h,
                                             bf16_t* __restrict__ hb) {
  const int i = blockIdx.x * blockDim.x + threadIdx.x;   // i < B*H
  const int b = i / H_, j = i % H_;
  const long o = (long)b * 3 * H_ + j;
  const float r  = sigmoidf_(gi[o] + gh[o]);
  const float z  = sigmoidf_(gi[o + H_] + gh[o + H_]);
  const float n  = tanhf(gi[o + 2 * H_] + r * gh[o + 2 * H_]);
  const float hv = (1.0f - z) * n + z * h[i];
  h[i] = hv;
  hb[i] = (bf16_t)hv;
}

// ---------------------------------------------------------------------------
// In-place log-softmax over V for row (b, t) of logp [B,T,V]; one block per b
// ---------------------------------------------------------------------------
__global__ __launch_bounds__(256) void k_logsoftmax(float* __restrict__ logp, int t) {
  const int b = blockIdx.x, tid = threadIdx.x;
  float* row = logp + ((long)b * T_ + t) * V_;
  __shared__ float red[256];

  float m = -1e30f;
  for (int v = tid; v < V_; v += 256) m = fmaxf(m, row[v]);
  red[tid] = m; __syncthreads();
  for (int s = 128; s > 0; s >>= 1) {
    if (tid < s) red[tid] = fmaxf(red[tid], red[tid + s]);
    __syncthreads();
  }
  m = red[0]; __syncthreads();

  float sum = 0.0f;
  for (int v = tid; v < V_; v += 256) sum += expf(row[v] - m);
  red[tid] = sum; __syncthreads();
  for (int s = 128; s > 0; s >>= 1) {
    if (tid < s) red[tid] += red[tid + s];
    __syncthreads();
  }
  const float shift = m + logf(red[0]);

  for (int v = tid; v < V_; v += 256) row[v] -= shift;
}

__global__ __launch_bounds__(256) void k_copy(const float* __restrict__ src,
                                              float* __restrict__ dst, int n) {
  int i = blockIdx.x * blockDim.x + threadIdx.x;
  if (i < n) dst[i] = src[i];
}

// ---------------------------------------------------------------------------
extern "C" void kernel_launch(void* const* d_in, const int* in_sizes, int n_in,
                              void* d_out, int out_size, void* d_ws, size_t ws_size,
                              hipStream_t stream) {
  (void)in_sizes; (void)n_in; (void)out_size; (void)ws_size;
  const float* enc   = (const float*)d_in[0];
  const float* h0    = (const float*)d_in[1];
  const int*   tgt   = (const int*)  d_in[2];
  const float* emb   = (const float*)d_in[3];
  const float* Wa    = (const float*)d_in[4];
  const float* b_wa  = (const float*)d_in[5];
  const float* Ua    = (const float*)d_in[6];
  const float* b_ua  = (const float*)d_in[7];
  const float* Va    = (const float*)d_in[8];
  const float* b_va  = (const float*)d_in[9];
  const float* W_ih  = (const float*)d_in[10];
  const float* W_hh  = (const float*)d_in[11];
  const float* b_ih  = (const float*)d_in[12];
  const float* b_hh  = (const float*)d_in[13];
  const float* W_out = (const float*)d_in[14];
  const float* b_out = (const float*)d_in[15];

  float* out      = (float*)d_out;
  float* logp     = out;                                    // [B,T,V]
  float* hfin     = out + (long)B_ * T_ * V_;               // [1,B,H]
  float* attn_out = hfin + (long)B_ * H_;                   // [B,T,S]

  // workspace carve-up (bf16 weights resident in L2 across all 32 steps)
  char* ws = (char*)d_ws;
  size_t off = 0;
  auto carve = [&](size_t bytes) -> char* {
    char* p = ws + off; off += (bytes + 255) & ~(size_t)255; return p;
  };
  bf16_t* Wa_b   = (bf16_t*)carve((size_t)H_ * H_ * 2);
  bf16_t* Ua_b   = (bf16_t*)carve((size_t)H_ * H_ * 2);
  bf16_t* Wih_b  = (bf16_t*)carve((size_t)3 * H_ * 2 * H_ * 2);
  bf16_t* Whh_b  = (bf16_t*)carve((size_t)3 * H_ * H_ * 2);
  bf16_t* Wout_b = (bf16_t*)carve((size_t)V_ * H_ * 2);
  bf16_t* enc_b  = (bf16_t*)carve((size_t)B_ * S_ * H_ * 2);
  float*  keys   = (float*) carve((size_t)B_ * S_ * H_ * 4);
  float*  h      = (float*) carve((size_t)B_ * H_ * 4);
  bf16_t* h_b    = (bf16_t*)carve((size_t)B_ * H_ * 2);
  float*  q      = (float*) carve((size_t)B_ * H_ * 4);
  bf16_t* x_b    = (bf16_t*)carve((size_t)B_ * 2 * H_ * 2);
  float*  gi     = (float*) carve((size_t)B_ * 3 * H_ * 4);
  float*  gh     = (float*) carve((size_t)B_ * 3 * H_ * 4);

  auto conv = [&](const float* s, bf16_t* d, long n) {
    k_f32_to_bf16<<<2048, 256, 0, stream>>>(s, d, n);
  };
  auto gemm = [&](const bf16_t* A, const bf16_t* W, const float* bias,
                  float* C, int M, int N, int K, long ldc) {
    int tiles = (M / 64) * (N / 16);        // one wave per 64x16 strip
    k_wmma_gemm<<<(tiles + 7) / 8, 256, 0, stream>>>(A, W, bias, C, M, N, K, ldc);
  };

  // -------- phase 0: weight quantization + keys_proj precompute ------------
  conv(Wa,    Wa_b,   (long)H_ * H_);
  conv(Ua,    Ua_b,   (long)H_ * H_);
  conv(W_ih,  Wih_b,  (long)3 * H_ * 2 * H_);
  conv(W_hh,  Whh_b,  (long)3 * H_ * H_);
  conv(W_out, Wout_b, (long)V_ * H_);
  conv(enc,   enc_b,  (long)B_ * S_ * H_);
  k_init_h<<<(B_ * H_) / 256, 256, 0, stream>>>(h0, h, h_b, B_ * H_);
  gemm(enc_b, Ua_b, b_ua, keys, B_ * S_, H_, H_, H_);      // [B*S,H]

  // -------- sequential decode over T ---------------------------------------
  for (int t = 0; t < T_; ++t) {
    gemm(h_b, Wa_b, b_wa, q, B_, H_, H_, H_);              // q = h Wa^T
    k_attn<<<B_, 256, 0, stream>>>(q, keys, enc, emb, tgt, Va, b_va,
                                   x_b, attn_out, t);
    gemm(x_b, Wih_b, b_ih, gi, B_, 3 * H_, 2 * H_, 3 * H_);
    gemm(h_b, Whh_b, b_hh, gh, B_, 3 * H_, H_, 3 * H_);
    k_gru<<<(B_ * H_) / 256, 256, 0, stream>>>(gi, gh, h, h_b);
    // logits straight into d_out slice [:, t, :]  (ldc = T*V)
    gemm(h_b, Wout_b, b_out, logp + (long)t * V_, B_, V_, H_, (long)T_ * V_);
    k_logsoftmax<<<B_, 256, 0, stream>>>(logp, t);
  }

  k_copy<<<(B_ * H_) / 256, 256, 0, stream>>>(h, hfin, B_ * H_);
}
